// MoeTransformerLayer_13932873908673
// MI455X (gfx1250) — compile-verified, matrix operand order checked by reading, not run on hardware
//
#include <hip/hip_runtime.h>

// ---------------- problem constants (match reference) ----------------
#define SLEN   2048
#define DMODEL 2048
#define NHEADS 16
#define HDIM   128
#define NB     2
#define NTOK   (NB * SLEN)     // 4096 tokens
#define NEXP   4

typedef __attribute__((ext_vector_type(16))) __bf16 v16bf;
typedef __attribute__((ext_vector_type(8)))  float  v8f;
typedef __attribute__((ext_vector_type(4)))  unsigned v4u;
typedef __attribute__((ext_vector_type(8)))  int    v8i;
typedef __attribute__((ext_vector_type(4)))  int    v4i;

// ---------------- TDM (Tensor Data Mover) support, feature-guarded ----------------
#if __has_builtin(__builtin_amdgcn_tensor_load_to_lds) && __has_builtin(__builtin_amdgcn_s_wait_tensorcnt)
#define USE_TDM 1
#endif

#ifdef USE_TDM
// Issue a 2D tensor DMA load (global -> LDS). Descriptor layout per
// CDNA5 ISA (D# group 0 + group 1). data_size = 2 bytes.
__device__ __forceinline__ void tdm_load_2d(unsigned lds_off, const void* gptr,
                                            unsigned tensor_d0, unsigned tensor_d1,
                                            unsigned tile_d0, unsigned tile_d1,
                                            unsigned long long d0_stride,
                                            unsigned pad_en, unsigned pad_interval,
                                            unsigned pad_amount) {
  const unsigned long long ga = (unsigned long long)(size_t)gptr;
  v4u g0;
  g0.x = 1u;                                                   // count=1 (valid user D#)
  g0.y = lds_off;                                              // lds_addr [63:32]
  g0.z = (unsigned)(ga & 0xffffffffu);                         // global_addr [95:64]
  g0.w = (unsigned)((ga >> 32) & 0x01ffffffu) | (2u << 30);    // global_addr hi + type=2
  const unsigned w0 = (1u << 16)                               // data_size = 2 bytes
                    | (pad_en << 20) | (pad_interval << 22) | (pad_amount << 25);
  const unsigned w1 = (tensor_d0 & 0xffffu) << 16;             // tensor_dim0[15:0] @ bits 63:48
  const unsigned w2 = (tensor_d0 >> 16) | ((tensor_d1 & 0xffffu) << 16);
  const unsigned w3 = (tensor_d1 >> 16) | (tile_d0 << 16);     // tile_dim0 @ bits 127:112
  const unsigned w4 = (tile_d1 & 0xffffu);                     // tile_dim1; tile_dim2=0
  const unsigned w5 = (unsigned)(d0_stride & 0xffffffffu);     // tensor_dim0_stride[31:0]
  const unsigned w6 = (unsigned)((d0_stride >> 32) & 0xffffu); // stride[47:32]; dim1_stride=0
  const v8i g1 = {(int)w0, (int)w1, (int)w2, (int)w3, (int)w4, (int)w5, (int)w6, 0};
  const v4i z4 = {0, 0, 0, 0};
#if __clang_major__ >= 23
  const v8i z8 = {0, 0, 0, 0, 0, 0, 0, 0};
  __builtin_amdgcn_tensor_load_to_lds(g0, g1, z4, z4, z8, 0);
#else
  __builtin_amdgcn_tensor_load_to_lds(g0, g1, z4, z4, 0);
#endif
}
#endif

// 16-bit WMMA A/B operand: for lane group g (=lane>>4), elements 0..7 hold
// K = g*8 + (0..7), elements 8..15 hold K = 16 + g*8 + (0..7) -> two 16B runs.
struct u4x2 { uint4 a, b; };
__device__ __forceinline__ v16bf frag_from(const __bf16* run0, const __bf16* run1) {
  u4x2 t;
  t.a = *(const uint4*)run0;
  t.b = *(const uint4*)run1;
  return __builtin_bit_cast(v16bf, t);
}

__device__ __forceinline__ unsigned pack2bf(float a, float b) {
  unsigned short x = __builtin_bit_cast(unsigned short, (__bf16)a);
  unsigned short y = __builtin_bit_cast(unsigned short, (__bf16)b);
  return (unsigned)x | ((unsigned)y << 16);
}

// ---------------- LayerNorm ----------------
__global__ void __launch_bounds__(256)
layernorm_kernel(const float* __restrict__ x, const float* __restrict__ w,
                 const float* __restrict__ b, float* __restrict__ y) {
  const int row = blockIdx.x;
  const float* xr = x + (size_t)row * DMODEL;
  float vals[8];
  float lsum = 0.f, lsq = 0.f;
#pragma unroll
  for (int i = 0; i < 8; ++i) {
    float v = xr[threadIdx.x + i * 256];
    vals[i] = v; lsum += v; lsq += v * v;
  }
  __shared__ float s1[256], s2[256];
  s1[threadIdx.x] = lsum; s2[threadIdx.x] = lsq;
  __syncthreads();
  for (int s = 128; s > 0; s >>= 1) {
    if (threadIdx.x < s) { s1[threadIdx.x] += s1[threadIdx.x + s]; s2[threadIdx.x] += s2[threadIdx.x + s]; }
    __syncthreads();
  }
  const float mean = s1[0] * (1.f / DMODEL);
  const float var  = s2[0] * (1.f / DMODEL) - mean * mean;
  const float inv  = rsqrtf(var + 1e-5f);
  float* yr = y + (size_t)row * DMODEL;
#pragma unroll
  for (int i = 0; i < 8; ++i) {
    int c = threadIdx.x + i * 256;
    yr[c] = (vals[i] - mean) * inv * w[c] + b[c];
  }
}

// ---------------- generic WMMA GEMM: C = act(A[MxK] * B[KxN] + bias) ----------------
// 128 threads = 4 waves in a 2x2 grid; block tile 64x64, K-step 32.
// MODE 0: store (acc+bias)        MODE 1: store gelu(acc+bias)
// MODE 2: C += rowscale[row*NEXP] * (acc+bias)   (gate-weighted expert accumulate)
template <int MODE>
__global__ void __launch_bounds__(128)
gemm_wmma_kernel(const float* __restrict__ A, const float* __restrict__ Bm,
                 const float* __restrict__ bias, float* __restrict__ C,
                 const float* __restrict__ rowscale, int M, int N, int K) {
  __shared__ __align__(16) __bf16 As[64][40];    // [m][k], 80B rows (16B aligned)
  __shared__ __align__(16) __bf16 Bst[64][40];   // TRANSPOSED: [n][k], 80B rows

  const int tid  = threadIdx.x;
  const int wave = tid >> 5;
  const int lane = tid & 31;
  const int g    = lane >> 4;
  const int ln16 = lane & 15;
  const int wm   = wave >> 1;     // wave row (2)
  const int wn   = wave & 1;      // wave col (2)
  const int m0 = blockIdx.y * 64;
  const int n0 = blockIdx.x * 64;

  v8f acc[2][2];
#pragma unroll
  for (int mt = 0; mt < 2; ++mt)
#pragma unroll
    for (int nt = 0; nt < 2; ++nt)
#pragma unroll
      for (int r = 0; r < 8; ++r) acc[mt][nt][r] = 0.f;

  const int ar = tid >> 1, ah = tid & 1;           // A stage: row, k-half
  const int bc4 = (tid & 15) * 4, bkq = tid >> 4;  // B stage: 4 cols, k-quad

  for (int k0 = 0; k0 < K; k0 += 32) {
    // ---- stage A tile 64x32 (row-major), bf16-packed uint2 stores ----
    {
      const float4* src = (const float4*)(A + (size_t)(m0 + ar) * K + k0 + ah * 16);
#pragma unroll
      for (int q = 0; q < 4; ++q) {
        float4 f = src[q];
        uint2 u; u.x = pack2bf(f.x, f.y); u.y = pack2bf(f.z, f.w);
        *(uint2*)&As[ar][ah * 16 + q * 4] = u;
      }
    }
    // ---- stage B tile 32x64 transposed into Bst[n][k] ----
    {
      float4 f[4];
#pragma unroll
      for (int dk = 0; dk < 4; ++dk)
        f[dk] = *(const float4*)(Bm + (size_t)(k0 + bkq * 4 + dk) * N + n0 + bc4);
#pragma unroll
      for (int cc = 0; cc < 4; ++cc) {
        const float v0 = (&f[0].x)[cc], v1 = (&f[1].x)[cc];
        const float v2 = (&f[2].x)[cc], v3 = (&f[3].x)[cc];
        uint2 u; u.x = pack2bf(v0, v1); u.y = pack2bf(v2, v3);
        *(uint2*)&Bst[bc4 + cc][bkq * 4] = u;
      }
    }
    if (k0 + 32 < K) {  // global_prefetch_b8 of next K tile
      __builtin_prefetch(A + (size_t)(m0 + ar) * K + (k0 + 32) + ah * 16, 0, 1);
      __builtin_prefetch(Bm + (size_t)(k0 + 32 + bkq * 4) * N + n0 + bc4, 0, 1);
    }
    __syncthreads();

    v16bf afrag[2], bfrag[2];
#pragma unroll
    for (int mt = 0; mt < 2; ++mt) {
      const int arow = wm * 32 + mt * 16 + ln16;
      afrag[mt] = frag_from(&As[arow][g * 8], &As[arow][16 + g * 8]);
    }
#pragma unroll
    for (int nt = 0; nt < 2; ++nt) {
      const int bcol = wn * 32 + nt * 16 + ln16;
      bfrag[nt] = frag_from(&Bst[bcol][g * 8], &Bst[bcol][16 + g * 8]);
    }
#pragma unroll
    for (int mt = 0; mt < 2; ++mt)
#pragma unroll
      for (int nt = 0; nt < 2; ++nt)
        acc[mt][nt] = __builtin_amdgcn_wmma_f32_16x16x32_bf16(
            false, afrag[mt], false, bfrag[nt], (short)0, acc[mt][nt], false, false);
    __syncthreads();
  }

#pragma unroll
  for (int mt = 0; mt < 2; ++mt)
#pragma unroll
    for (int nt = 0; nt < 2; ++nt) {
      const int col = n0 + wn * 32 + nt * 16 + ln16;
      const float bval = bias[col];
#pragma unroll
      for (int r = 0; r < 8; ++r) {
        const int row = m0 + wm * 32 + mt * 16 + g * 8 + r;
        float v = acc[mt][nt][r] + bval;
        if (MODE == 1) v = 0.5f * v * (1.f + erff(v * 0.70710678118654752f));  // exact GELU
        const size_t idx = (size_t)row * N + col;
        if (MODE == 2) C[idx] += rowscale[(size_t)row * NEXP] * v;
        else           C[idx] = v;
      }
    }
}

// ---------------- RoPE + reshape; V is written TRANSPOSED [B,H,HD,S] ----------------
__global__ void __launch_bounds__(64)
rope_kernel(const float* __restrict__ qf, const float* __restrict__ kf,
            const float* __restrict__ vf, __bf16* __restrict__ qb,
            __bf16* __restrict__ kb, __bf16* __restrict__ vbt) {
  const int bid = blockIdx.x;            // over B*S*H
  const int h = bid % NHEADS;
  const int s = (bid / NHEADS) % SLEN;
  const int b = bid / (NHEADS * SLEN);
  const int j = threadIdx.x;             // 0..63 (pair index)
  const size_t in  = ((size_t)b * SLEN + s) * DMODEL + (size_t)h * HDIM;
  const size_t out = (((size_t)b * NHEADS + h) * SLEN + s) * HDIM;
  const size_t bhH = (size_t)(b * NHEADS + h) * HDIM;

  const float invf = __powf(10000.f, -(float)(2 * j) / (float)HDIM);
  const float ang = (float)s * invf;
  const float c = __cosf(ang), sn = __sinf(ang);

  float q1 = qf[in + j], q2 = qf[in + j + 64];
  qb[out + j]      = (__bf16)(q1 * c - q2 * sn);
  qb[out + j + 64] = (__bf16)(q2 * c + q1 * sn);
  float k1 = kf[in + j], k2 = kf[in + j + 64];
  kb[out + j]      = (__bf16)(k1 * c - k2 * sn);
  kb[out + j + 64] = (__bf16)(k2 * c + k1 * sn);
  vbt[(bhH + j) * SLEN + s]      = (__bf16)vf[in + j];
  vbt[(bhH + j + 64) * SLEN + s] = (__bf16)vf[in + j + 64];
}

// ---------------- flash attention (causal), 1 wave per 16 query rows ----------------
__global__ void __launch_bounds__(32)
flash_attn_kernel(const __bf16* __restrict__ qb, const __bf16* __restrict__ kb,
                  const __bf16* __restrict__ vbt, float* __restrict__ out) {
  const int q0 = blockIdx.x * 16;
  const int bh = blockIdx.y;            // b*NHEADS + h
  const int lane = threadIdx.x;
  const int g = lane >> 4, ln16 = lane & 15;

  __shared__ __align__(16) __bf16 Kb[32][HDIM];   // [kv][hd], 256B rows
  __shared__ __align__(16) __bf16 Vt[HDIM][40];   // [hd][kv], 80B rows (32 + 8 pad)
  __shared__ __align__(16) __bf16 Pb[16][40];     // [qrow][kv], bf16 probs

#ifdef USE_TDM
  const unsigned kb_off = (unsigned)(size_t)&Kb[0][0];
  const unsigned vt_off = (unsigned)(size_t)&Vt[0][0];
#endif

  // Q tile 16x128 in registers, A-operand layout (4 chunks of K=32)
  const __bf16* Qbase = qb + ((size_t)bh * SLEN + (q0 + ln16)) * HDIM;
  v16bf qfrag[4];
#pragma unroll
  for (int kc = 0; kc < 4; ++kc)
    qfrag[kc] = frag_from(Qbase + kc * 32 + g * 8, Qbase + kc * 32 + 16 + g * 8);

  v8f oacc[8];
  float rmax[8], rsum[8];
#pragma unroll
  for (int nt = 0; nt < 8; ++nt)
#pragma unroll
    for (int r = 0; r < 8; ++r) oacc[nt][r] = 0.f;
#pragma unroll
  for (int r = 0; r < 8; ++r) { rmax[r] = -1e30f; rsum[r] = 0.f; }

  const float sscale = 0.08838834764831845f;  // 1/sqrt(128)

  for (int kv0 = 0; kv0 < q0 + 16; kv0 += 32) {
#ifdef USE_TDM
    // ---- TDM: K tile 32 rows x 128 bf16 (contiguous), V tile 128 rows x 32 bf16
    //      with LDS row padding 16 DWORDs data + 4 DWORDs pad (codes 3 / 3) ----
    tdm_load_2d(kb_off, kb + ((size_t)bh * SLEN + kv0) * HDIM,
                /*tensor*/ HDIM, 32, /*tile*/ HDIM, 32, /*stride*/ (unsigned long long)HDIM,
                /*pad*/ 0, 0, 0);
    tdm_load_2d(vt_off, vbt + ((size_t)bh * HDIM) * SLEN + kv0,
                /*tensor*/ 32, HDIM, /*tile*/ 32, HDIM, /*stride*/ (unsigned long long)SLEN,
                /*pad*/ 1, 3, 3);
    __builtin_amdgcn_s_wait_tensorcnt(0);
#else
    // ---- manual staging fallback: b128 copies ----
    {
      const uint4* ks = (const uint4*)(kb + ((size_t)bh * SLEN + kv0 + lane) * HDIM);
      uint4* kd = (uint4*)(&Kb[lane][0]);
#pragma unroll
      for (int i = 0; i < 16; ++i) kd[i] = ks[i];
    }
#pragma unroll
    for (int d = 0; d < 4; ++d) {
      const int hd = lane * 4 + d;
      const uint4* vs = (const uint4*)(vbt + ((size_t)bh * HDIM + hd) * SLEN + kv0);
      uint4* vd = (uint4*)(&Vt[hd][0]);
      vd[0] = vs[0]; vd[1] = vs[1]; vd[2] = vs[2]; vd[3] = vs[3];
    }
#endif
    __syncthreads();

    // ---- scores: two 16x16 tiles over 32 kv columns, K=128 in 4 WMMA steps ----
    v8f s0, s1;
#pragma unroll
    for (int r = 0; r < 8; ++r) { s0[r] = 0.f; s1[r] = 0.f; }
#pragma unroll
    for (int kc = 0; kc < 4; ++kc) {
      const v16bf b0 = frag_from(&Kb[ln16][kc * 32 + g * 8], &Kb[ln16][kc * 32 + 16 + g * 8]);
      const v16bf b1 = frag_from(&Kb[16 + ln16][kc * 32 + g * 8], &Kb[16 + ln16][kc * 32 + 16 + g * 8]);
      s0 = __builtin_amdgcn_wmma_f32_16x16x32_bf16(false, qfrag[kc], false, b0, (short)0, s0, false, false);
      s1 = __builtin_amdgcn_wmma_f32_16x16x32_bf16(false, qfrag[kc], false, b1, (short)0, s1, false, false);
    }

    // ---- online softmax over the 32 new columns (fp32) ----
    const int c0 = kv0 + ln16, c1 = kv0 + 16 + ln16;
#pragma unroll
    for (int r = 0; r < 8; ++r) {
      const int qrow = q0 + g * 8 + r;
      const float a0 = (c0 <= qrow) ? s0[r] * sscale : -1e30f;
      const float a1 = (c1 <= qrow) ? s1[r] * sscale : -1e30f;
      float tm = fmaxf(a0, a1);
      tm = fmaxf(tm, __shfl_xor(tm, 1, 32));
      tm = fmaxf(tm, __shfl_xor(tm, 2, 32));
      tm = fmaxf(tm, __shfl_xor(tm, 4, 32));
      tm = fmaxf(tm, __shfl_xor(tm, 8, 32));
      const float nm = fmaxf(rmax[r], tm);
      const float corr = __expf(rmax[r] - nm);
      const float p0 = (c0 <= qrow) ? __expf(a0 - nm) : 0.f;
      const float p1 = (c1 <= qrow) ? __expf(a1 - nm) : 0.f;
      float ps = p0 + p1;
      ps += __shfl_xor(ps, 1, 32);
      ps += __shfl_xor(ps, 2, 32);
      ps += __shfl_xor(ps, 4, 32);
      ps += __shfl_xor(ps, 8, 32);
      rsum[r] = rsum[r] * corr + ps;
      rmax[r] = nm;
#pragma unroll
      for (int nt = 0; nt < 8; ++nt) oacc[nt][r] *= corr;
      Pb[g * 8 + r][ln16]      = (__bf16)p0;
      Pb[g * 8 + r][16 + ln16] = (__bf16)p1;
    }
    __syncthreads();

    // ---- P(16x32) @ V(32x128): all fragments are b128 pairs ----
    const v16bf pfrag = frag_from(&Pb[ln16][g * 8], &Pb[ln16][16 + g * 8]);
#pragma unroll
    for (int nt = 0; nt < 8; ++nt) {
      const int vcol = nt * 16 + ln16;
      const v16bf vfr = frag_from(&Vt[vcol][g * 8], &Vt[vcol][16 + g * 8]);
      oacc[nt] = __builtin_amdgcn_wmma_f32_16x16x32_bf16(false, pfrag, false, vfr, (short)0, oacc[nt], false, false);
    }
    __syncthreads();
  }

  // ---- normalize and write [B,S,H*HD] fp32 ----
  const int bb = bh / NHEADS, hh = bh % NHEADS;
  float rinv[8];
#pragma unroll
  for (int r = 0; r < 8; ++r) rinv[r] = 1.f / rsum[r];
#pragma unroll
  for (int nt = 0; nt < 8; ++nt)
#pragma unroll
    for (int r = 0; r < 8; ++r) {
      const int q = q0 + g * 8 + r;
      out[((size_t)bb * SLEN + q) * DMODEL + hh * HDIM + nt * 16 + ln16] = oacc[nt][r] * rinv[r];
    }
}

// ---------------- gate: softmax -> top-2 -> renormalize -> dense combine [T,E] ----------------
__global__ void __launch_bounds__(32)
gate_topk_kernel(const float* __restrict__ x, const float* __restrict__ wg,
                 float* __restrict__ comb) {
  const int t = blockIdx.x;
  const int lane = threadIdx.x;
  const float* xr = x + (size_t)t * DMODEL;
  float a[NEXP] = {0.f, 0.f, 0.f, 0.f};
  for (int d = lane; d < DMODEL; d += 32) {
    const float xv = xr[d];
#pragma unroll
    for (int e = 0; e < NEXP; ++e) a[e] += xv * wg[(size_t)d * NEXP + e];
  }
#pragma unroll
  for (int e = 0; e < NEXP; ++e)
    for (int m = 16; m >= 1; m >>= 1) a[e] += __shfl_xor(a[e], m, 32);
  if (lane == 0) {
    float mx = fmaxf(fmaxf(a[0], a[1]), fmaxf(a[2], a[3]));
    float p[NEXP], s = 0.f;
#pragma unroll
    for (int e = 0; e < NEXP; ++e) { p[e] = __expf(a[e] - mx); s += p[e]; }
#pragma unroll
    for (int e = 0; e < NEXP; ++e) p[e] /= s;
    int i1 = 0;
    for (int e = 1; e < NEXP; ++e) if (p[e] > p[i1]) i1 = e;
    int i2 = -1;
    for (int e = 0; e < NEXP; ++e) { if (e == i1) continue; if (i2 < 0 || p[e] > p[i2]) i2 = e; }
    const float e1 = __expf(p[i1]), e2 = __expf(p[i2]);  // softmax over the top-k values
    const float den = e1 + e2;
    float o[NEXP] = {0.f, 0.f, 0.f, 0.f};
    o[i1] = e1 / den; o[i2] = e2 / den;
#pragma unroll
    for (int e = 0; e < NEXP; ++e) comb[(size_t)t * NEXP + e] = o[e];
  }
}

// ---------------- residual: h = x + attn_proj; d_out initialized to h ----------------
__global__ void __launch_bounds__(256)
add_resid_kernel(const float* __restrict__ x, const float* __restrict__ a,
                 float* __restrict__ h, float* __restrict__ out, int n) {
  const int i = blockIdx.x * 256 + threadIdx.x;
  if (i < n) { const float v = x[i] + a[i]; h[i] = v; out[i] = v; }
}

// ---------------- host-side orchestration ----------------
extern "C" void kernel_launch(void* const* d_in, const int* in_sizes, int n_in,
                              void* d_out, int out_size, void* d_ws, size_t ws_size,
                              hipStream_t stream) {
  const float* x    = (const float*)d_in[0];
  const float* wq   = (const float*)d_in[1];
  const float* bq   = (const float*)d_in[2];
  const float* wk   = (const float*)d_in[3];
  const float* bk   = (const float*)d_in[4];
  const float* wv   = (const float*)d_in[5];
  const float* bv   = (const float*)d_in[6];
  const float* wo   = (const float*)d_in[7];
  const float* bo   = (const float*)d_in[8];
  const float* ln1w = (const float*)d_in[9];
  const float* ln1b = (const float*)d_in[10];
  const float* ln2w = (const float*)d_in[11];
  const float* ln2b = (const float*)d_in[12];
  const float* wg   = (const float*)d_in[13];
  const float* we1  = (const float*)d_in[14];
  const float* be1  = (const float*)d_in[15];
  const float* we2  = (const float*)d_in[16];
  const float* be2  = (const float*)d_in[17];
  float* out = (float*)d_out;

  char* ws = (char*)d_ws;
  const size_t F32BUF = (size_t)NTOK * DMODEL * sizeof(float);   // 32 MiB
  const size_t BFBUF  = (size_t)NTOK * DMODEL * sizeof(__bf16);  // 16 MiB
  float*  xln  = (float*)(ws);
  float*  qf   = (float*)(ws + 1 * F32BUF);   // also reused: attn out, MoE hmid
  float*  kf   = (float*)(ws + 2 * F32BUF);   // also reused: attn projection
  float*  vf   = (float*)(ws + 3 * F32BUF);   // also reused: h (residual)
  __bf16* qb   = (__bf16*)(ws + 4 * F32BUF);
  __bf16* kb   = (__bf16*)(ws + 4 * F32BUF + 1 * BFBUF);
  __bf16* vbt  = (__bf16*)(ws + 4 * F32BUF + 2 * BFBUF);  // transposed [B,H,HD,S]
  float*  comb = (float*)(ws + 4 * F32BUF + 3 * BFBUF);

  const dim3 ggrid(DMODEL / 64, NTOK / 64);   // (32, 64) output tiles

  // 1) LN1
  layernorm_kernel<<<NTOK, 256, 0, stream>>>(x, ln1w, ln1b, xln);
  // 2) QKV projections
  gemm_wmma_kernel<0><<<ggrid, 128, 0, stream>>>(xln, wq, bq, qf, nullptr, NTOK, DMODEL, DMODEL);
  gemm_wmma_kernel<0><<<ggrid, 128, 0, stream>>>(xln, wk, bk, kf, nullptr, NTOK, DMODEL, DMODEL);
  gemm_wmma_kernel<0><<<ggrid, 128, 0, stream>>>(xln, wv, bv, vf, nullptr, NTOK, DMODEL, DMODEL);
  // 3) RoPE + bf16 reshape (V transposed)
  rope_kernel<<<NB * SLEN * NHEADS, 64, 0, stream>>>(qf, kf, vf, qb, kb, vbt);
  // 4) causal flash attention -> qf ([B,S,H*HD] fp32)
  {
    dim3 fgrid(SLEN / 16, NB * NHEADS);
    flash_attn_kernel<<<fgrid, 32, 0, stream>>>(qb, kb, vbt, qf);
  }
  // 5) output projection -> kf
  gemm_wmma_kernel<0><<<ggrid, 128, 0, stream>>>(qf, wo, bo, kf, nullptr, NTOK, DMODEL, DMODEL);
  // 6) h = x + attn; d_out = h
  {
    const int n = NTOK * DMODEL;
    add_resid_kernel<<<(n + 255) / 256, 256, 0, stream>>>(x, kf, vf, out, n);
  }
  // 7) LN2 -> xln
  layernorm_kernel<<<NTOK, 256, 0, stream>>>(vf, ln2w, ln2b, xln);
  // 8) gate / top-2 combine weights
  gate_topk_kernel<<<NTOK, 32, 0, stream>>>(xln, wg, comb);
  // 9) experts: hmid = gelu(xln @ we1[e] + be1[e]); d_out += comb[:,e] * (hmid @ we2[e] + be2[e])
  for (int e = 0; e < NEXP; ++e) {
    gemm_wmma_kernel<1><<<ggrid, 128, 0, stream>>>(
        xln, we1 + (size_t)e * DMODEL * DMODEL, be1 + (size_t)e * DMODEL, qf, nullptr,
        NTOK, DMODEL, DMODEL);
    gemm_wmma_kernel<2><<<ggrid, 128, 0, stream>>>(
        qf, we2 + (size_t)e * DMODEL * DMODEL, be2 + (size_t)e * DMODEL, out,
        comb + e, NTOK, DMODEL, DMODEL);
  }
}
